// NodeUpdate_9990093930530
// MI455X (gfx1250) — matrix operand order repaired and checked by reading, not compiled, not run on hardware
//
#include <hip/hip_runtime.h>

typedef __attribute__((ext_vector_type(16))) _Float16 v16h;
typedef __attribute__((ext_vector_type(8)))  _Float16 v8h;
typedef __attribute__((ext_vector_type(4)))  _Float16 v4h;
typedef __attribute__((ext_vector_type(8)))  float    v8f;
typedef __attribute__((ext_vector_type(4)))  float    v4f;

#define HN   128      // node hidden
#define KD   256      // fan_in = Hn + He
#define NO   256      // 2*Hn
#define BN_EPS 1e-5f

// ---------------------------------------------------------------------------
// Zero a float region (stats + agg are laid out contiguously in ws)
__global__ void k_zero(float* __restrict__ p, size_t count) {
    size_t idx    = (size_t)blockIdx.x * blockDim.x + threadIdx.x;
    size_t stride = (size_t)gridDim.x * blockDim.x;
    for (; idx < count; idx += stride) p[idx] = 0.f;
}

// int64 indices -> int32 scratch (reference i is int64)
__global__ void k_cvt_idx(const long long* __restrict__ i64, int* __restrict__ i32, int E) {
    int e = blockIdx.x * 256 + threadIdx.x;
    if (e < E) i32[e] = (int)i64[e];
}

// W1 [K=256][N=256] fp32 row-major  ->  W1T [N][K] f16 (B operand, L2-resident 128KB)
__global__ void k_cvt_w1(const float* __restrict__ W1, _Float16* __restrict__ w1t) {
    int n = blockIdx.x, k = threadIdx.x;
    w1t[(size_t)n * KD + k] = (_Float16)W1[(size_t)k * NO + n];
}

// ---------------------------------------------------------------------------
// Fused gather+concat -> f16 GEMM (WMMA f32_16x16x32_f16) -> zT f16 [NO][E]
// plus inline per-column sum / sumsq for BatchNorm1 (BN is shift-invariant,
// so b1 is mathematically irrelevant and skipped).
// Block: 256 threads = 8 waves; tile 64 rows x 256 cols; waves 4(M) x 2(N).
__global__ __launch_bounds__(256)
void k_gemm_stats(const float* __restrict__ node_emb,
                  const float* __restrict__ edge_emb,
                  const int*   __restrict__ idx,
                  const _Float16* __restrict__ w1t,   // [NO][KD] f16
                  _Float16* __restrict__ zT,          // [NO][E]  f16
                  float* __restrict__ colsum, float* __restrict__ colsq,
                  int E)
{
    constexpr int LDSS = KD + 8;                 // pad to break bank alignment
    __shared__ _Float16 sA[64 * LDSS];
    __shared__ float sCS[NO];
    __shared__ float sCQ[NO];

    const int tid = threadIdx.x;
    const int e0  = blockIdx.x * 64;

    if (tid < NO) { sCS[tid] = 0.f; sCQ[tid] = 0.f; }

    // Stage A tile (gather node_emb[i[e]] || edge_emb[e]) as f16 in LDS.
    for (int t = tid; t < 64 * 64; t += 256) {
        const int row = t >> 6;
        const int q   = t & 63;                  // 4-column group (cols 4q..4q+3)
        const int e   = e0 + row;
        v4f v;
        if (e < E) {
            const float* src = (q < 32)
                ? node_emb + (size_t)idx[e] * HN + q * 4
                : edge_emb + (size_t)e   * HN + (q - 32) * 4;
            v = *(const v4f*)src;
        } else {
            v = (v4f){0.f, 0.f, 0.f, 0.f};
        }
        v4h h = { (_Float16)v.x, (_Float16)v.y, (_Float16)v.z, (_Float16)v.w };
        *(v4h*)(sA + row * LDSS + q * 4) = h;
    }
    __syncthreads();

    const int lane = tid & 31;
    const int w    = tid >> 5;
    const int wm   = w >> 1;                     // 0..3 : 16-row strip
    const int wn   = w & 1;                      // 0..1 : 128-col strip
    const int lr   = lane & 15;
    const int half = lane >> 4;

    const _Float16* a_row  = sA + (wm * 16 + lr) * LDSS;
    const int       n_base = wn * 128;

    v8f acc[8] = {};

    for (int k0 = 0; k0 < KD; k0 += 32) {
        // A 16x32 f16 fragment (ISA layout: lane-half selects K-offset 0/8)
        v8h alo = *(const v8h*)(a_row + k0 + half * 8);
        v8h ahi = *(const v8h*)(a_row + k0 + 16 + half * 8);
        v16h a  = __builtin_shufflevector(alo, ahi,
                     0,1,2,3,4,5,6,7,8,9,10,11,12,13,14,15);
        #pragma unroll
        for (int t = 0; t < 8; ++t) {
            // B 32x16 f16 fragment: lane half h -> K 8h..8h+7 and 16+8h..16+8h+7
            const _Float16* brow =
                w1t + (size_t)(n_base + t * 16 + lr) * KD + k0 + half * 8;
            v8h blo = *(const v8h*)(brow);
            v8h bhi = *(const v8h*)(brow + 16);
            v16h b  = __builtin_shufflevector(blo, bhi,
                         0,1,2,3,4,5,6,7,8,9,10,11,12,13,14,15);
            acc[t] = __builtin_amdgcn_wmma_f32_16x16x32_f16(
                         false, a, false, b, (short)0, acc[t], false, false);
        }
    }

    // Epilogue: C layout lane -> (n = lr, m = half*8 + r). Store zT + stats.
    #pragma unroll
    for (int t = 0; t < 8; ++t) {
        const int n  = n_base + t * 16 + lr;
        const int eb = e0 + wm * 16 + half * 8;  // 8 consecutive rows
        v8h h;
        float s = 0.f, q = 0.f;
        #pragma unroll
        for (int r = 0; r < 8; ++r) {
            float v = acc[t][r];
            h[r] = (_Float16)v;
            s += v; q += v * v;
        }
        if (eb + 8 <= E) {
            *(v8h*)(zT + (size_t)n * E + eb) = h;   // 16B contiguous store
        } else {
            for (int r = 0; r < 8; ++r)
                if (eb + r < E) zT[(size_t)n * E + eb + r] = h[r];
        }
        s += __shfl_xor(s, 16);                  // fold the two M-halves
        q += __shfl_xor(q, 16);
        if (half == 0) { atomicAdd(&sCS[n], s); atomicAdd(&sCQ[n], q); }
    }
    __syncthreads();
    if (tid < NO) {
        unsafeAtomicAdd(&colsum[tid], sCS[tid]);
        unsafeAtomicAdd(&colsq[tid],  sCQ[tid]);
    }
}

// ---------------------------------------------------------------------------
// Fold stats into scale/shift:  y = z*scale + shift  ==  gamma*(z-mu)*rstd+beta
__global__ void k_finalize(const float* __restrict__ colsum, const float* __restrict__ colsq,
                           const float* __restrict__ gamma,  const float* __restrict__ beta,
                           float* __restrict__ scale, float* __restrict__ shift, float inv_cnt) {
    const int n   = threadIdx.x;
    const float m = colsum[n] * inv_cnt;
    const float v = colsq[n] * inv_cnt - m * m;
    const float s = gamma[n] * rsqrtf(v + BN_EPS);
    scale[n] = s;
    shift[n] = beta[n] - m * s;
}

// ---------------------------------------------------------------------------
// BN1 + gate + segment_sum. One block.x per output column n (0..127);
// column-major zT makes both reads fully coalesced.
__global__ __launch_bounds__(256)
void k_scatter(const _Float16* __restrict__ zT, const int* __restrict__ idx,
               const float* __restrict__ scale, const float* __restrict__ shift,
               float* __restrict__ agg, int E)
{
    const int   n  = blockIdx.x;
    const float sf = scale[n],       hf = shift[n];
    const float sc = scale[n + HN],  hc = shift[n + HN];
    const _Float16* zf = zT + (size_t)n * E;
    const _Float16* zc = zT + (size_t)(n + HN) * E;
    int e      = blockIdx.y * 256 + threadIdx.x;
    int stride = gridDim.y * 256;
    for (; e < E; e += stride) {
        float f   = (float)zf[e] * sf + hf;
        float c   = (float)zc[e] * sc + hc;
        float msg = (1.f / (1.f + __expf(-f))) * tanhf(c);
        unsafeAtomicAdd(&agg[(size_t)idx[e] * HN + n], msg);
    }
}

// ---------------------------------------------------------------------------
// Column stats of agg [N,128] for BN2.
__global__ __launch_bounds__(256)
void k_reduce2(const float* __restrict__ agg,
               float* __restrict__ cs, float* __restrict__ cq, size_t total)
{
    __shared__ float ls[HN], lq[HN];
    const int tid = threadIdx.x;
    if (tid < HN) { ls[tid] = 0.f; lq[tid] = 0.f; }
    __syncthreads();
    float s = 0.f, q = 0.f;
    size_t idx    = (size_t)blockIdx.x * 256 + tid;
    size_t stride = (size_t)gridDim.x * 256;     // multiple of 128 -> n fixed/thread
    for (; idx < total; idx += stride) {
        float v = agg[idx];
        s += v; q += v * v;
    }
    atomicAdd(&ls[tid & (HN - 1)], s);
    atomicAdd(&lq[tid & (HN - 1)], q);
    __syncthreads();
    if (tid < HN) { unsafeAtomicAdd(&cs[tid], ls[tid]); unsafeAtomicAdd(&cq[tid], lq[tid]); }
}

// out = tanh(node_emb + BN2(agg))
__global__ void k_final(const float* __restrict__ node_emb, const float* __restrict__ agg,
                        const float* __restrict__ scale, const float* __restrict__ shift,
                        float* __restrict__ out, size_t total)
{
    size_t idx    = (size_t)blockIdx.x * blockDim.x + threadIdx.x;
    size_t stride = (size_t)gridDim.x * blockDim.x;
    for (; idx < total; idx += stride) {
        int n = (int)(idx & (HN - 1));
        out[idx] = tanhf(node_emb[idx] + agg[idx] * scale[n] + shift[n]);
    }
}

// ---------------------------------------------------------------------------
extern "C" void kernel_launch(void* const* d_in, const int* in_sizes, int n_in,
                              void* d_out, int out_size, void* d_ws, size_t ws_size,
                              hipStream_t stream)
{
    (void)n_in; (void)out_size; (void)ws_size;
    const float*     node_emb = (const float*)d_in[0];
    const float*     edge_emb = (const float*)d_in[1];
    const long long* i64      = (const long long*)d_in[2];   // reference: int64
    const float*     W1       = (const float*)d_in[3];
    // d_in[4] = b1: exactly cancelled by BatchNorm shift-invariance
    const float*     gamma1   = (const float*)d_in[5];
    const float*     beta1    = (const float*)d_in[6];
    const float*     gamma2   = (const float*)d_in[7];
    const float*     beta2    = (const float*)d_in[8];

    const int N = in_sizes[0] / HN;
    const int E = in_sizes[2];

    // Workspace layout (stats + agg contiguous so one zero pass covers them)
    char* ws = (char*)d_ws;
    size_t off = 0;
    auto alloc = [&](size_t bytes) { void* p = ws + off; off = (off + bytes + 255) & ~(size_t)255; return p; };
    float* colsum1 = (float*)(ws);                 // 256
    float* colsq1  = colsum1 + NO;                 // 256
    float* colsum2 = colsq1 + NO;                  // 128
    float* colsq2  = colsum2 + HN;                 // 128
    float* agg     = colsq2 + HN;                  // N*128  (contiguous zero region)
    off = ((NO * 2 + HN * 2 + (size_t)N * HN) * sizeof(float) + 255) & ~(size_t)255;
    float*     scale1 = (float*)alloc(NO * sizeof(float));
    float*     shift1 = (float*)alloc(NO * sizeof(float));
    float*     scale2 = (float*)alloc(HN * sizeof(float));
    float*     shift2 = (float*)alloc(HN * sizeof(float));
    int*       i32    = (int*)alloc((size_t)E * sizeof(int));
    _Float16*  w1t    = (_Float16*)alloc((size_t)NO * KD * sizeof(_Float16));
    _Float16*  zT     = (_Float16*)alloc((size_t)NO * E * sizeof(_Float16));

    const size_t zero_cnt = (size_t)NO * 2 + (size_t)HN * 2 + (size_t)N * HN;
    k_zero<<<2048, 256, 0, stream>>>(colsum1, zero_cnt);
    k_cvt_idx<<<(E + 255) / 256, 256, 0, stream>>>(i64, i32, E);
    k_cvt_w1<<<NO, KD, 0, stream>>>(W1, w1t);

    k_gemm_stats<<<(E + 63) / 64, 256, 0, stream>>>(
        node_emb, edge_emb, i32, w1t, zT, colsum1, colsq1, E);

    k_finalize<<<1, NO, 0, stream>>>(colsum1, colsq1, gamma1, beta1,
                                     scale1, shift1, 1.f / (float)E);
    k_scatter<<<dim3(HN, 512), 256, 0, stream>>>(zT, i32, scale1, shift1, agg, E);
    k_reduce2<<<1024, 256, 0, stream>>>(agg, colsum2, colsq2, (size_t)N * HN);
    k_finalize<<<1, HN, 0, stream>>>(colsum2, colsq2, gamma2, beta2,
                                     scale2, shift2, 1.f / (float)N);
    k_final<<<4096, 256, 0, stream>>>(node_emb, agg, scale2, shift2,
                                      (float*)d_out, (size_t)N * HN);
}
